// WideExpertMoE_63900523430547
// MI455X (gfx1250) — matrix operand order repaired and checked by reading, not compiled, not run on hardware
//
#include <hip/hip_runtime.h>
#include <math.h>

// Problem constants (match reference)
#define Bc 2
#define Sc 2048
#define Hc 1024
#define Ec 16
#define Nc (Bc * Sc)   // 4096 tokens

// GEMM tiling
#define BM 32          // bucket entries (token-slots) per block tile
#define BN 128         // output dims per block tile
#define WN 64          // output dims per wave

typedef __attribute__((ext_vector_type(16))) __bf16 bf16x16;
typedef __attribute__((ext_vector_type(8)))  float  f32x8;

__device__ __forceinline__ unsigned short f2bf(float f) {
    unsigned u = __float_as_uint(f);
    u += 0x7FFFu + ((u >> 16) & 1u);   // round-to-nearest-even
    return (unsigned short)(u >> 16);
}

// ---------------- zero expert counts ----------------
__global__ void WideExpertMoE_zero_counts(int* __restrict__ counts) {
    if (threadIdx.x < Ec) counts[threadIdx.x] = 0;
}

// ---------------- fp32 -> bf16 conversion ----------------
__global__ void WideExpertMoE_cvt(const float* __restrict__ src,
                                  unsigned short* __restrict__ dst, int n4) {
    int i = blockIdx.x * blockDim.x + threadIdx.x;
    if (i >= n4) return;
    float4 v = ((const float4*)src)[i];
    ushort4 o;
    o.x = f2bf(v.x); o.y = f2bf(v.y); o.z = f2bf(v.z); o.w = f2bf(v.w);
    ((ushort4*)dst)[i] = o;
}

// ---------------- gate: logits -> softmax -> top2 -> bucket scatter ----------------
// Bucket entry packs (token, slot): v = 2*token + slot, so the GEMM can write
// its contribution to Y[v*H + d] with no atomics (each v written exactly once).
__global__ void WideExpertMoE_route(const float* __restrict__ tokens,
                                    const float* __restrict__ gate_w,
                                    int* __restrict__ counts,
                                    int* __restrict__ btok,
                                    float* __restrict__ bw) {
    __shared__ float lg[16][17];
    const int tx = threadIdx.x & 15;   // expert
    const int ty = threadIdx.x >> 4;   // token within block
    const int n  = blockIdx.x * 16 + ty;

    const float4* xp = (const float4*)(tokens + (size_t)n * Hc);
    const float4* gp = (const float4*)(gate_w + (size_t)tx * Hc);
    float acc = 0.f;
    for (int h = 0; h < Hc / 4; ++h) {
        float4 a = xp[h], g = gp[h];
        acc += a.x * g.x + a.y * g.y + a.z * g.z + a.w * g.w;
    }
    lg[ty][tx] = acc;
    __syncthreads();

    if (tx == 0) {
        float m = lg[ty][0];
        for (int e = 1; e < Ec; ++e) m = fmaxf(m, lg[ty][e]);
        float s = 0.f;
        for (int e = 0; e < Ec; ++e) s += __expf(lg[ty][e] - m);
        // top-1 (first index wins ties, matching lax.top_k)
        int e0 = 0; float l0 = lg[ty][0];
        for (int e = 1; e < Ec; ++e) if (lg[ty][e] > l0) { l0 = lg[ty][e]; e0 = e; }
        int e1 = -1; float l1 = -INFINITY;
        for (int e = 0; e < Ec; ++e) if (e != e0 && lg[ty][e] > l1) { l1 = lg[ty][e]; e1 = e; }
        float w0 = __expf(l0 - m) / s;
        float w1 = __expf(l1 - m) / s;
        int p0 = atomicAdd(&counts[e0], 1); btok[e0 * Nc + p0] = 2 * n + 0; bw[e0 * Nc + p0] = w0;
        int p1 = atomicAdd(&counts[e1], 1); btok[e1 * Nc + p1] = 2 * n + 1; bw[e1 * Nc + p1] = w1;
    }
}

// ---------------- grouped expert GEMM: Y[v] = gate_w * relu(x W^T + b) ----------------
__global__ void WideExpertMoE_gemm(const unsigned short* __restrict__ Xb,
                                   const unsigned short* __restrict__ Wb,
                                   const float* __restrict__ expert_b,
                                   const int* __restrict__ counts,
                                   const int* __restrict__ btok,
                                   const float* __restrict__ bw,
                                   float* __restrict__ Y) {
    const int e     = blockIdx.z;
    const int cnt   = counts[e];
    const int mbase = blockIdx.y * BM;
    if (mbase >= cnt) return;
    const int dblock = blockIdx.x * BN;

    __shared__ int   s_v[BM];    // packed (2*token + slot)
    __shared__ float s_w[BM];
    const int tid = threadIdx.x;
    if (tid < BM) {
        int idx = mbase + tid;
        bool ok = idx < cnt;
        s_v[tid] = ok ? btok[e * Nc + idx] : 0;
        s_w[tid] = ok ? bw[e * Nc + idx]   : 0.f;
    }
    __syncthreads();

    const int wave  = tid >> 5;      // 0..3
    const int lane  = tid & 31;
    const int wm    = wave >> 1;     // token-half of block tile
    const int wn    = wave & 1;      // d-half of block tile
    const int l16   = lane & 15;
    const int khalf = lane >> 4;     // 0/1

    // A fragment (16x32 bf16): lane holds row l16; K runs of 8 at
    // k0 + khalf*8 and k0 + khalf*8 + 16
    const int tokA = s_v[wm * 16 + l16] >> 1;
    const unsigned short* arow = Xb + (size_t)tokA * Hc;

    // B fragment (32x16 bf16): lane holds column l16; K runs of 8 at
    // k0 + khalf*8 and k0 + khalf*8 + 16
    const unsigned short* brow[4];
#pragma unroll
    for (int sub = 0; sub < 4; ++sub) {
        int d = dblock + wn * WN + sub * 16 + l16;
        brow[sub] = Wb + ((size_t)e * Hc + d) * Hc;
    }

    f32x8 acc[4] = {};
    union Frag { uint4 q[2]; bf16x16 v; };

    const int kA = khalf * 8;
    const int kB = khalf * 8;
    for (int k0 = 0; k0 < Hc; k0 += 32) {
        // prefetch next K tile (speculative; emits global_prefetch_b8)
        __builtin_prefetch((const void*)(arow + k0 + 32 + kA), 0, 1);
        __builtin_prefetch((const void*)(brow[0] + k0 + 32 + kB), 0, 1);

        Frag a;
        a.q[0] = *(const uint4*)(arow + k0 + kA);
        a.q[1] = *(const uint4*)(arow + k0 + kA + 16);
#pragma unroll
        for (int sub = 0; sub < 4; ++sub) {
            Frag b;
            b.q[0] = *(const uint4*)(brow[sub] + k0 + kB);
            b.q[1] = *(const uint4*)(brow[sub] + k0 + kB + 16);
            acc[sub] = __builtin_amdgcn_wmma_f32_16x16x32_bf16(
                false, a.v, false, b.v, (short)0, acc[sub], false, false);
        }
    }

    // Epilogue: D element (m,n): n = lane%16 (column d), m = 8*(lane/16) + r.
    // Plain stores into per-slot buffer: each (token,slot,d) written once.
#pragma unroll
    for (int sub = 0; sub < 4; ++sub) {
        const int d      = dblock + wn * WN + sub * 16 + l16;
        const float bias = expert_b[e * Hc + d];
#pragma unroll
        for (int r = 0; r < 8; ++r) {
            const int ml = wm * 16 + khalf * 8 + r;
            if (mbase + ml < cnt) {
                float v = fmaxf(acc[sub][r] + bias, 0.f);
                Y[(size_t)s_v[ml] * Hc + d] = s_w[ml] * v;
            }
        }
    }
}

// ---------------- combine the two expert slots per token ----------------
__global__ void WideExpertMoE_combine(const float* __restrict__ Y,
                                      float* __restrict__ out, int n4) {
    int i = blockIdx.x * blockDim.x + threadIdx.x;
    if (i >= n4) return;
    int elem = i * 4;
    int n = elem >> 10;           // / Hc
    int d = elem & (Hc - 1);
    const float4 a = *(const float4*)(Y + ((size_t)(2 * n + 0) * Hc + d));
    const float4 b = *(const float4*)(Y + ((size_t)(2 * n + 1) * Hc + d));
    float4 o;
    o.x = a.x + b.x; o.y = a.y + b.y; o.z = a.z + b.z; o.w = a.w + b.w;
    ((float4*)out)[i] = o;
}

extern "C" void kernel_launch(void* const* d_in, const int* in_sizes, int n_in,
                              void* d_out, int out_size, void* d_ws, size_t ws_size,
                              hipStream_t stream) {
    const float* tokens   = (const float*)d_in[0];
    const float* gate_w   = (const float*)d_in[1];
    const float* expert_w = (const float*)d_in[2];
    const float* expert_b = (const float*)d_in[3];
    float* out = (float*)d_out;

    // Workspace layout
    char* ws = (char*)d_ws;
    unsigned short* Xb = (unsigned short*)ws;                                  //  8 MiB
    unsigned short* Wb = (unsigned short*)(ws + (size_t) 8 * 1024 * 1024);     // 32 MiB
    float*          Yb = (float*)         (ws + (size_t)40 * 1024 * 1024);     // 32 MiB (N*2*H f32)
    int*   counts = (int*)(ws + (size_t)72 * 1024 * 1024);
    int*   btok   = counts + 64;                                               // E*N ints
    float* bww    = (float*)(btok + Ec * Nc);                                  // E*N floats

    WideExpertMoE_zero_counts<<<1, 64, 0, stream>>>(counts);

    WideExpertMoE_cvt<<<(Nc * Hc / 4 + 255) / 256, 256, 0, stream>>>(tokens, Xb, Nc * Hc / 4);
    WideExpertMoE_cvt<<<(Ec * Hc * Hc / 4 + 255) / 256, 256, 0, stream>>>(expert_w, Wb, Ec * Hc * Hc / 4);

    WideExpertMoE_route<<<Nc / 16, 256, 0, stream>>>(tokens, gate_w, counts, btok, bww);

    dim3 grid(Hc / BN, Nc / BM, Ec);   // worst-case M tiles; empty tiles early-exit
    WideExpertMoE_gemm<<<grid, 128, 0, stream>>>(Xb, Wb, expert_b, counts, btok, bww, Yb);

    WideExpertMoE_combine<<<(Nc * Hc / 4 + 255) / 256, 256, 0, stream>>>(Yb, out, Nc * Hc / 4);
}
